// CENet_Transformer_61710090109632
// MI455X (gfx1250) — compile-verified, matrix operand order checked by reading, not run on hardware
//
#include <hip/hip_runtime.h>
#include <hip/hip_bf16.h>

// ---------------------------------------------------------------------------
// CDNA5 (gfx1250) deformable-attention encoder, WMMA f16 path.
// ---------------------------------------------------------------------------

typedef __attribute__((ext_vector_type(16))) _Float16 v16h;
typedef __attribute__((ext_vector_type(8)))  float    v8f;
typedef _Float16 half_t;

// Problem constants (fixed by the reference setup).
constexpr int Bc  = 4;
constexpr int Hc  = 128;
constexpr int Wc  = 128;
constexpr int Lqc = Hc * Wc;        // 16384
constexpr int BLq = Bc * Lqc;       // 65536 rows
constexpr int DMc = 128;
constexpr int NHc = 4;
constexpr int NPc = 4;
constexpr int HDc = 32;
constexpr int FFc = 512;
constexpr int Lc  = 2;

// ---------------------------------------------------------------------------
// Weight pre-pack: fp32 [K,N] row-major -> fp16 WMMA B-fragment layout.
// One 32x16 (KxN) tile per block of 32 lanes; lane l (l<16) holds column
// N=n0+l, K rows k0..k0+15; lane l>=16 holds the same column, K rows
// k0+16..k0+31 (contiguous k = 16h + i, the same order the GEMM loads A in).
// Tile storage: tile*(32*16) + lane*16 + i   (contiguous per lane).
// ---------------------------------------------------------------------------
__global__ void pack_weight_kernel(const float* __restrict__ W,
                                   half_t* __restrict__ Wp, int K, int N) {
  const int tile  = blockIdx.x;
  const int lane  = threadIdx.x;          // 0..31
  const int nt16  = N >> 4;
  const int kt    = tile / nt16;
  const int nt    = tile - kt * nt16;
  const int col   = nt * 16 + (lane & 15);
  const int kbase = kt * 32 + (lane >> 4) * 16;
  half_t* dst = Wp + (size_t)tile * 512 + lane * 16;
#pragma unroll
  for (int i = 0; i < 16; ++i)
    dst[i] = (half_t)W[(size_t)(kbase + i) * N + col];
}

// ---------------------------------------------------------------------------
// fp32 (+ optional fp32 add) -> fp16 conversion (qin = query+pos, src).
// ---------------------------------------------------------------------------
__global__ void add_convert_kernel(const float* __restrict__ a,
                                   const float* __restrict__ b,
                                   half_t* __restrict__ out, int n) {
  int i = blockIdx.x * blockDim.x + threadIdx.x;
  if (i >= n) return;
  float v = a[i];
  if (b) v += b[i];
  out[i] = (half_t)v;
}

// ---------------------------------------------------------------------------
// WMMA GEMM: out[M,N] = A[M,K] (f16, row-major) x Wp (f16, pre-packed)
//            + bias, optional ReLU; fp32 and/or fp16 stores.
// Block = 128 threads = 4 waves; each wave computes a 16 x (16*NTILES)
// strip; A fragment reused across NTILES WMMAs per K step. Everything is
// compile-time static (NTILES, K, epilogue) so no accumulator phis / spills.
// ---------------------------------------------------------------------------
template <int NTILES, int KC, bool RELU, bool OUTF, bool OUTH>
__global__ __launch_bounds__(128) void gemm_wmma_f16_kernel(
    const half_t* __restrict__ A, const half_t* __restrict__ Wp,
    const float* __restrict__ bias, float* __restrict__ outF,
    half_t* __restrict__ outH, int N) {
  const int lane = threadIdx.x & 31;
  const int wave = threadIdx.x >> 5;
  const int m0 = (blockIdx.y * 4 + wave) * 16;
  const int n0 = blockIdx.x * (16 * NTILES);

  // A-fragment addressing (16-bit A 16x32 layout): lanes 0-15 -> K 0..15,
  // lanes 16-31 -> K 16..31, row M = lane&15.
  const int arow = m0 + (lane & 15);
  const int akb  = (lane >> 4) * 16;
  constexpr int KTILES = KC / 32;
  const int nt16 = N >> 4;

  v8f acc[NTILES] = {};

  const half_t* aptr = A + (size_t)arow * KC + akb;
#pragma unroll
  for (int kt = 0; kt < KTILES; ++kt) {
    v16h a = *(const v16h*)(aptr + kt * 32);
    if (kt + 1 < KTILES)
      __builtin_prefetch(aptr + (kt + 1) * 32, 0, 3);   // global_prefetch_b8
    const half_t* wb =
        Wp + ((size_t)kt * nt16 + (n0 >> 4)) * 512 + lane * 16;
#pragma unroll
    for (int t = 0; t < NTILES; ++t) {
      v16h b = *(const v16h*)(wb + t * 512);
      acc[t] = __builtin_amdgcn_wmma_f32_16x16x32_f16(
          /*neg_a=*/false, a, /*neg_b=*/false, b,
          /*c_mod=*/(short)0, acc[t], /*reuse_a=*/false, /*reuse_b=*/false);
    }
  }

  // C/D layout: lane l, VGPR r -> C[m0 + r + (l>=16)*8][n0 + (l&15)].
  const int colb = n0 + (lane & 15);
  const int rowb = m0 + (lane >> 4) * 8;
#pragma unroll
  for (int t = 0; t < NTILES; ++t) {
    const int col = colb + t * 16;
    const float bv = bias[col];
#pragma unroll
    for (int r = 0; r < 8; ++r) {
      float v = acc[t][r] + bv;
      if (RELU) v = fmaxf(v, 0.0f);
      const size_t o = (size_t)(rowb + r) * N + col;
      if (OUTF) outF[o] = v;
      if (OUTH) outH[o] = (half_t)v;
    }
  }
}

// ---------------------------------------------------------------------------
// Softmax over NP=4 contiguous attention logits per (b,q,head); in place.
// ---------------------------------------------------------------------------
__global__ void softmax4_kernel(float* __restrict__ aw, int ngroups) {
  int t = blockIdx.x * blockDim.x + threadIdx.x;
  if (t >= ngroups) return;
  float* p = aw + (size_t)t * 4;
  float a0 = p[0], a1 = p[1], a2 = p[2], a3 = p[3];
  float m = fmaxf(fmaxf(a0, a1), fmaxf(a2, a3));
  float e0 = expf(a0 - m), e1 = expf(a1 - m), e2 = expf(a2 - m),
        e3 = expf(a3 - m);
  float inv = 1.0f / (e0 + e1 + e2 + e3);
  p[0] = e0 * inv; p[1] = e1 * inv; p[2] = e2 * inv; p[3] = e3 * inv;
}

// ---------------------------------------------------------------------------
// Deformable bilinear sampling + attention-weighted reduce over points.
// One thread per (b, q, head). Pixel coords simplify on this grid:
// (rx + ox/W)*W - 0.5 = qx + ox   (align_corners=False, ref at cell centers).
// ---------------------------------------------------------------------------
__global__ void deform_sample_kernel(const float* __restrict__ value,
                                     const float* __restrict__ so,
                                     const float* __restrict__ aw,
                                     half_t* __restrict__ attn_h) {
  int t = blockIdx.x * blockDim.x + threadIdx.x;
  if (t >= BLq * NHc) return;
  const int h  = t & (NHc - 1);
  const int bq = t >> 2;
  const int b  = bq >> 14;             // Lq = 16384
  const int q  = bq & (Lqc - 1);
  const int qy = q >> 7;               // W = 128
  const int qx = q & (Wc - 1);

  const float* vb  = value + ((size_t)b * Lqc) * DMc + h * HDc;
  const float* sop = so + (size_t)bq * (NHc * NPc * 2) + h * (NPc * 2);
  const float* awp = aw + (size_t)bq * (NHc * NPc) + h * NPc;

  float acc[HDc];
#pragma unroll
  for (int c = 0; c < HDc; ++c) acc[c] = 0.0f;

#pragma unroll
  for (int p = 0; p < NPc; ++p) {
    const float w = awp[p];
    const float x = (float)qx + sop[p * 2 + 0];
    const float y = (float)qy + sop[p * 2 + 1];
    const float xf = floorf(x), yf = floorf(y);
    const int x0 = (int)xf, y0 = (int)yf;
    const float wx = x - xf, wy = y - yf;
    const float cw[4] = {(1.f - wx) * (1.f - wy), wx * (1.f - wy),
                         (1.f - wx) * wy,         wx * wy};
    const int cx[4] = {x0, x0 + 1, x0, x0 + 1};
    const int cy[4] = {y0, y0, y0 + 1, y0 + 1};
#pragma unroll
    for (int k = 0; k < 4; ++k) {
      const int xi = cx[k], yi = cy[k];
      if (xi >= 0 && xi < Wc && yi >= 0 && yi < Hc) {
        const float4* s4 =
            (const float4*)(vb + (size_t)(yi * Wc + xi) * DMc);
        const float wk = w * cw[k];
#pragma unroll
        for (int c4 = 0; c4 < HDc / 4; ++c4) {
          const float4 v4 = s4[c4];
          acc[c4 * 4 + 0] += wk * v4.x;
          acc[c4 * 4 + 1] += wk * v4.y;
          acc[c4 * 4 + 2] += wk * v4.z;
          acc[c4 * 4 + 3] += wk * v4.w;
        }
      }
    }
  }
  half_t* outp = attn_h + (size_t)bq * DMc + h * HDc;
#pragma unroll
  for (int c = 0; c < HDc; ++c) outp[c] = (half_t)acc[c];
}

// ---------------------------------------------------------------------------
// Residual add + LayerNorm over DM=128, one block (128 thr) per row.
// outF may alias resid. Optional fp16 copy for the next GEMM.
// ---------------------------------------------------------------------------
__global__ void resid_ln_kernel(const float* __restrict__ resid,
                                const float* __restrict__ delta,
                                const float* __restrict__ g,
                                const float* __restrict__ bta,
                                float* __restrict__ outF,
                                half_t* __restrict__ outH) {
  const int row = blockIdx.x;
  const int c   = threadIdx.x;
  const size_t idx = (size_t)row * DMc + c;
  const float x = resid[idx] + delta[idx];

  __shared__ float red[DMc];
  red[c] = x;
  __syncthreads();
  for (int s = DMc / 2; s > 0; s >>= 1) {
    if (c < s) red[c] += red[c + s];
    __syncthreads();
  }
  const float mean = red[0] * (1.0f / DMc);
  __syncthreads();
  const float d = x - mean;
  red[c] = d * d;
  __syncthreads();
  for (int s = DMc / 2; s > 0; s >>= 1) {
    if (c < s) red[c] += red[c + s];
    __syncthreads();
  }
  const float var = red[0] * (1.0f / DMc);
  const float y = d * rsqrtf(var + 1e-5f) * g[c] + bta[c];
  outF[idx] = y;
  if (outH) outH[idx] = (half_t)y;
}

// ---------------------------------------------------------------------------
// Host orchestration.
// ---------------------------------------------------------------------------
extern "C" void kernel_launch(void* const* d_in, const int* in_sizes, int n_in,
                              void* d_out, int out_size, void* d_ws,
                              size_t ws_size, hipStream_t stream) {
  (void)in_sizes; (void)n_in; (void)out_size; (void)ws_size;

  const float* query = (const float*)d_in[0];
  const float* src   = (const float*)d_in[1];
  const float* pos   = (const float*)d_in[2];
  const float* so_w  = (const float*)d_in[3];
  const float* so_b  = (const float*)d_in[4];
  const float* aw_w  = (const float*)d_in[5];
  const float* aw_b  = (const float*)d_in[6];
  const float* vp_w  = (const float*)d_in[7];
  const float* vp_b  = (const float*)d_in[8];
  const float* op_w  = (const float*)d_in[9];
  const float* op_b  = (const float*)d_in[10];
  const float* ln1_g = (const float*)d_in[11];
  const float* ln1_b = (const float*)d_in[12];
  const float* l1_w  = (const float*)d_in[13];
  const float* l1_b  = (const float*)d_in[14];
  const float* l2_w  = (const float*)d_in[15];
  const float* l2_b  = (const float*)d_in[16];
  const float* ln2_g = (const float*)d_in[17];
  const float* ln2_b = (const float*)d_in[18];
  float* q_out = (float*)d_out;   // running query, fp32

  // ---- workspace carve-out ----
  char* base = (char*)d_ws;
  size_t off = 0;
  auto carve = [&](size_t bytes) {
    char* p = base + off;
    off = (off + bytes + 255) & ~(size_t)255;
    return p;
  };
  half_t* src_h   = (half_t*)carve((size_t)BLq * DMc * 2);
  half_t* qin_h   = (half_t*)carve((size_t)BLq * DMc * 2);
  half_t* q_h     = (half_t*)carve((size_t)BLq * DMc * 2);
  half_t* attn_h  = (half_t*)carve((size_t)BLq * DMc * 2);
  half_t* ffn_h   = (half_t*)carve((size_t)BLq * FFc * 2);
  float*  so_out  = (float*)carve((size_t)BLq * 32 * 4);
  float*  aw_buf  = (float*)carve((size_t)BLq * 16 * 4);
  float*  value   = (float*)carve((size_t)BLq * DMc * 4);
  float*  scratchF= (float*)carve((size_t)BLq * DMc * 4);
  half_t* so_p[Lc], *aw_p[Lc], *vp_p[Lc], *op_p[Lc], *l1_p[Lc], *l2_p[Lc];
  for (int l = 0; l < Lc; ++l) {
    so_p[l] = (half_t*)carve((size_t)DMc * 32  * 2);
    aw_p[l] = (half_t*)carve((size_t)DMc * 16  * 2);
    vp_p[l] = (half_t*)carve((size_t)DMc * DMc * 2);
    op_p[l] = (half_t*)carve((size_t)DMc * DMc * 2);
    l1_p[l] = (half_t*)carve((size_t)DMc * FFc * 2);
    l2_p[l] = (half_t*)carve((size_t)FFc * DMc * 2);
  }

  // ---- pack all weights to fp16 fragment layout ----
  auto pack = [&](const float* W, half_t* Wp, int K, int N) {
    dim3 grid((K / 32) * (N / 16)), block(32);
    hipLaunchKernelGGL(pack_weight_kernel, grid, block, 0, stream, W, Wp, K, N);
  };
  for (int l = 0; l < Lc; ++l) {
    pack(so_w + (size_t)l * DMc * 32,  so_p[l], DMc, 32);
    pack(aw_w + (size_t)l * DMc * 16,  aw_p[l], DMc, 16);
    pack(vp_w + (size_t)l * DMc * DMc, vp_p[l], DMc, DMc);
    pack(op_w + (size_t)l * DMc * DMc, op_p[l], DMc, DMc);
    pack(l1_w + (size_t)l * DMc * FFc, l1_p[l], DMc, FFc);
    pack(l2_w + (size_t)l * FFc * DMc, l2_p[l], FFc, DMc);
  }

  // ---- running query = input query; src -> fp16 once ----
  hipMemcpyAsync(q_out, query, (size_t)BLq * DMc * 4,
                 hipMemcpyDeviceToDevice, stream);
  {
    const int n = BLq * DMc;
    hipLaunchKernelGGL(add_convert_kernel, dim3((n + 255) / 256), dim3(256), 0,
                       stream, src, (const float*)nullptr, src_h, n);
  }

  const dim3 gblock(128);
  const dim3 grow(1, BLq / 64);                   // N <= 64 (1 block col)
  const dim3 g128(2, BLq / 64);                   // N = 128
  const dim3 g512(8, BLq / 64);                   // N = 512

  for (int l = 0; l < Lc; ++l) {
    // qin = query + pos  (fp16 for GEMM A)
    {
      const int n = BLq * DMc;
      hipLaunchKernelGGL(add_convert_kernel, dim3((n + 255) / 256), dim3(256),
                         0, stream, q_out, pos, qin_h, n);
    }
    // sampling offsets [N=32], attention logits [N=16], value proj [N=128]
    hipLaunchKernelGGL((gemm_wmma_f16_kernel<2, DMc, false, true, false>),
                       grow, gblock, 0, stream, qin_h, so_p[l],
                       so_b + (size_t)l * 32, so_out, (half_t*)nullptr, 32);
    hipLaunchKernelGGL((gemm_wmma_f16_kernel<1, DMc, false, true, false>),
                       grow, gblock, 0, stream, qin_h, aw_p[l],
                       aw_b + (size_t)l * 16, aw_buf, (half_t*)nullptr, 16);
    hipLaunchKernelGGL((gemm_wmma_f16_kernel<4, DMc, false, true, false>),
                       g128, gblock, 0, stream, src_h, vp_p[l],
                       vp_b + (size_t)l * DMc, value, (half_t*)nullptr, DMc);
    // softmax over NP per (b,q,head)
    {
      const int ng = BLq * NHc;
      hipLaunchKernelGGL(softmax4_kernel, dim3((ng + 255) / 256), dim3(256), 0,
                         stream, aw_buf, ng);
    }
    // deformable bilinear gather + weighted reduce -> attn (fp16)
    {
      const int nt = BLq * NHc;
      hipLaunchKernelGGL(deform_sample_kernel, dim3((nt + 255) / 256),
                         dim3(256), 0, stream, value, so_out, aw_buf, attn_h);
    }
    // output projection, residual + LN1 (fp16 copy feeds FFN)
    hipLaunchKernelGGL((gemm_wmma_f16_kernel<4, DMc, false, true, false>),
                       g128, gblock, 0, stream, attn_h, op_p[l],
                       op_b + (size_t)l * DMc, scratchF, (half_t*)nullptr, DMc);
    hipLaunchKernelGGL(resid_ln_kernel, dim3(BLq), dim3(DMc), 0, stream, q_out,
                       scratchF, ln1_g + (size_t)l * DMc,
                       ln1_b + (size_t)l * DMc, q_out, q_h);
    // FFN: 128 -> 512 (ReLU, fp16 out) -> 128, residual + LN2
    hipLaunchKernelGGL((gemm_wmma_f16_kernel<4, DMc, true, false, true>),
                       g512, gblock, 0, stream, q_h, l1_p[l],
                       l1_b + (size_t)l * FFc, (float*)nullptr, ffn_h, FFc);
    hipLaunchKernelGGL((gemm_wmma_f16_kernel<4, FFc, false, true, false>),
                       g128, gblock, 0, stream, ffn_h, l2_p[l],
                       l2_b + (size_t)l * DMc, scratchF, (half_t*)nullptr, DMc);
    hipLaunchKernelGGL(resid_ln_kernel, dim3(BLq), dim3(DMc), 0, stream, q_out,
                       scratchF, ln2_g + (size_t)l * DMc,
                       ln2_b + (size_t)l * DMc, q_out, (half_t*)nullptr);
  }
}
